// LayoutEncoder_10118942949779
// MI455X (gfx1250) — matrix operand order repaired and checked by reading, not compiled, not run on hardware
//
#include <hip/hip_runtime.h>

typedef _Float16 half_t;
typedef __attribute__((ext_vector_type(16))) _Float16 v16h;
typedef __attribute__((ext_vector_type(8)))  float    v8f;

#define CBN_EPS 1e-5f

__device__ __forceinline__ float sigf(float x) { return 1.0f / (1.0f + __expf(-x)); }

// ---------------------------------------------------------------------------
// Weight pack: OIHW f32 -> A-fragment-ordered f16.
// K order of the GEMM is (kh, kw, ic). Layout (halves):
//   ap[(((mt*KH + kh)*KW + kw)*nCk + ck)*512 + lane*16 + r*2 + h]
// holds A element (m = lane&15, k_in_chunk = (lane>>4)*16 + r*2 + h)
// i.e. the v16h A fragment in the contiguous k order that the B staging also uses.
// ---------------------------------------------------------------------------
__global__ __launch_bounds__(256) void pack_a_kernel(
    const float* __restrict__ w, half_t* __restrict__ ap,
    int Cout, int Cin, int KH, int KW)
{
  long total = (long)Cout * Cin * KH * KW / 2;
  long idx = (long)blockIdx.x * blockDim.x + threadIdx.x;
  if (idx >= total) return;
  const int nCk = Cin >> 5;
  int r    = (int)(idx & 7);
  long rest = idx >> 3;
  int lane = (int)(rest & 31); rest >>= 5;
  int ck   = (int)(rest % nCk); rest /= nCk;
  int kw   = (int)(rest % KW);  rest /= KW;
  int kh   = (int)(rest % KH);  rest /= KH;
  int mt   = (int)rest;
  int m  = lane & 15, hi = lane >> 4;
  int ka = hi * 16 + (r << 1);
  int oc = mt * 16 + m;
  int ic = ck * 32 + ka;
  long sbase = (((long)oc * Cin + ic) * KH + kh) * KW + kw;
  long dbase = ((((long)mt * KH + kh) * KW + kw) * nCk + ck) * 512 + lane * 16 + r * 2;
  ap[dbase]     = (half_t)w[sbase];
  ap[dbase + 1] = (half_t)w[sbase + (long)KH * KW];   // ic+1
}

// ---------------------------------------------------------------------------
// Implicit-GEMM conv, LDS-staged, v_wmma_f32_16x16x32_f16.
// Block: 256 threads = 8 waves -> macro-tile 32 (out-ch) x 128 (positions).
// wave w: mSub = w&1 (M subtile), nSub = w>>1 owns N-tiles {2*nSub, 2*nSub+1}.
// Per (kh,kw,ic-chunk):
//   - 256 threads cooperatively stage the 32ch x 128pos B chunk into LDS in
//     B-fragment order (16 independent-address loads -> one clause, one wait),
//   - each wave reads one packed-A fragment (sequential global b128 x2) and
//     two B fragments (ds b128 x2 each), issuing 2 WMMAs (A reused).
// Requires: Cout % 32 == 0, Cin % 32 == 0, Npos % 128 == 0 (true for all
// layers of this model).  EXEC is all-1s at every WMMA.
// ---------------------------------------------------------------------------
template <bool IN_F32>
__global__ __launch_bounds__(256) void wmma_conv_kernel(
    const void* __restrict__ in_v, const half_t* __restrict__ apack,
    const float* __restrict__ bias, float* __restrict__ out,
    int NO, int Cin, int Hin, int Win,
    int Cout, int Hout, int Wout,
    int KH, int KW, int stride, int pad)
{
  __shared__ alignas(32) half_t bstage[8 * 32 * 16];   // 8 KB

  const int t    = threadIdx.x;
  const int lane = t & 31;
  const int wv   = t >> 5;
  const int mSub = wv & 1;
  const int nSub = wv >> 1;            // 0..3

  const int  tilesM2 = Cout >> 5;
  const long blk     = blockIdx.x;
  const int  blockM  = (int)(blk % tilesM2);
  const long blockN  = blk / tilesM2;
  const long n0      = blockN * 128;

  const int HWo = Hout * Wout;
  const int HWi = Hin * Win;
  const int nCk = Cin >> 5;

  // ---- loader mapping: this thread services position n0 + (t&127) ---------
  const int  nld_i = t & 127;
  const long nld   = n0 + nld_i;
  const int  lo_o  = (int)(nld / HWo);
  const int  lrem  = (int)(nld % HWo);
  const int  loy   = lrem / Wout;
  const int  lox   = lrem % Wout;
  const int  chBase = t >> 7;          // 0..1 ; channels chBase + 2*i
  const int  ntile  = nld_i >> 4;      // 0..7
  const int  nlow   = nld_i & 15;

  // ---- compute mapping -----------------------------------------------------
  const int tileM = blockM * 2 + mSub;
  const int hi    = lane >> 4;
  const int nl    = lane & 15;

  const half_t* inh = (const half_t*)in_v;
  const float*  inf = (const float*)in_v;

  const half_t* ap     = apack + (long)tileM * KH * KW * nCk * 512 + (long)lane * 16;
  const half_t* bread0 = &bstage[(((nSub * 2 + 0) * 32) + lane) * 16];
  const half_t* bread1 = &bstage[(((nSub * 2 + 1) * 32) + lane) * 16];

  v8f acc0 = {0.f, 0.f, 0.f, 0.f, 0.f, 0.f, 0.f, 0.f};
  v8f acc1 = {0.f, 0.f, 0.f, 0.f, 0.f, 0.f, 0.f, 0.f};

  const long chStride = (long)2 * HWi;   // loader channel step (elements)
  const long ckStride = (long)32 * HWi;  // chunk step (elements)

  // Uniform, loop-invariant channel offsets (SGPRs): independent load addrs.
  long coff[16];
#pragma unroll
  for (int i = 0; i < 16; ++i) coff[i] = (long)i * chStride;

  for (int kh = 0; kh < KH; ++kh) {
    const int liy = loy * stride + kh - pad;
    for (int kw = 0; kw < KW; ++kw) {
      const int  lix    = lox * stride + kw - pad;
      const bool lvalid = (liy >= 0) && (liy < Hin) && (lix >= 0) && (lix < Win);
      const long lbase  = (long)lo_o * Cin * HWi + (long)liy * Win + lix;
      const float*  pf = inf + lbase + (long)chBase * HWi;
      const half_t* ph = inh + lbase + (long)chBase * HWi;

      for (int ck = 0; ck < nCk; ++ck) {
        __syncthreads();
        if (lvalid) {
          if (IN_F32) {
            float tmp[16];
#pragma unroll
            for (int i = 0; i < 16; ++i) tmp[i] = pf[coff[i]];
            int ch = chBase;
#pragma unroll
            for (int i = 0; i < 16; ++i) {
              bstage[((ntile * 32) + nlow + (ch & 16)) * 16 + (ch & 15)] = (half_t)tmp[i];
              ch += 2;
            }
          } else {
            half_t tmp[16];
#pragma unroll
            for (int i = 0; i < 16; ++i) tmp[i] = ph[coff[i]];
            int ch = chBase;
#pragma unroll
            for (int i = 0; i < 16; ++i) {
              bstage[((ntile * 32) + nlow + (ch & 16)) * 16 + (ch & 15)] = tmp[i];
              ch += 2;
            }
          }
        } else {
          int ch = chBase;
#pragma unroll
          for (int i = 0; i < 16; ++i) {
            bstage[((ntile * 32) + nlow + (ch & 16)) * 16 + (ch & 15)] = (half_t)0.0f;
            ch += 2;
          }
        }
        __syncthreads();
        v16h a  = *(const v16h*)ap;
        v16h b0 = *(const v16h*)bread0;
        v16h b1 = *(const v16h*)bread1;
        __builtin_prefetch(ap + 512, 0, 1);               // next packed-A chunk
        acc0 = __builtin_amdgcn_wmma_f32_16x16x32_f16(false, a, false, b0,
                                                      (short)0, acc0, false, false);
        acc1 = __builtin_amdgcn_wmma_f32_16x16x32_f16(false, a, false, b1,
                                                      (short)0, acc1, false, false);
        ap += 512;
        pf += ckStride; ph += ckStride;
      }
    }
  }

#pragma unroll
  for (int j = 0; j < 2; ++j) {
    const long nw  = n0 + (nSub * 2 + j) * 16 + nl;
    const int  o   = (int)(nw / HWo);
    const int  rem = (int)(nw % HWo);
    const int  oy  = rem / Wout;
    const int  ox  = rem % Wout;
    const long obase = (long)o * Cout * HWo + (long)oy * Wout + ox;
    const v8f& acc = j ? acc1 : acc0;
#pragma unroll
    for (int r = 0; r < 8; ++r) {
      const int oc = tileM * 16 + r + 8 * hi;
      float v = acc[r];
      if (bias) v += bias[oc];
      out[obase + (long)oc * HWo] = v;
    }
  }
}

// Per-channel mean/var over (NO, HW): one block per channel.
__global__ __launch_bounds__(256) void channel_stats_kernel(
    const float* __restrict__ x, float* __restrict__ mean, float* __restrict__ var,
    int NO, int C, int HW)
{
  __shared__ float ss[256], ss2[256];
  const int c = blockIdx.x;
  float s = 0.f, s2 = 0.f;
  for (int o = 0; o < NO; ++o) {
    const float* p = x + ((long)o * C + c) * HW;
    for (int j = threadIdx.x; j < HW; j += 256) {
      float v = p[j];
      s += v; s2 += v * v;
    }
  }
  ss[threadIdx.x] = s; ss2[threadIdx.x] = s2;
  __syncthreads();
  for (int off = 128; off > 0; off >>= 1) {
    if (threadIdx.x < off) {
      ss[threadIdx.x]  += ss[threadIdx.x + off];
      ss2[threadIdx.x] += ss2[threadIdx.x + off];
    }
    __syncthreads();
  }
  if (threadIdx.x == 0) {
    float inv = 1.f / ((float)NO * (float)HW);
    float m   = ss[0] * inv;
    mean[c]   = m;
    var[c]    = fmaxf(ss2[0] * inv - m * m, 0.f);
  }
}

// Conditional BN apply (+ optional leaky relu), emits f16 for the next conv.
__global__ __launch_bounds__(256) void cbn_apply_kernel(
    const float* __restrict__ x, const float* __restrict__ mean,
    const float* __restrict__ var, const float* __restrict__ embed,
    const int* __restrict__ objs, half_t* __restrict__ out,
    int C, int HW, long total, int do_lrelu)
{
  long idx = (long)blockIdx.x * blockDim.x + threadIdx.x;
  if (idx >= total) return;
  int c   = (int)((idx / HW) % C);
  int o   = (int)(idx / ((long)C * HW));
  int cls = objs[o];
  float g  = embed[(long)cls * 2 * C + c];
  float be = embed[(long)cls * 2 * C + C + c];
  float v  = g * (x[idx] - mean[c]) * rsqrtf(var[c] + CBN_EPS) + be;
  if (do_lrelu) v = (v >= 0.f) ? v : 0.2f * v;
  out[idx] = (half_t)v;
}

// Build [x ; h] channel concat (f16). h == nullptr means h = 0 (t = 0).
__global__ __launch_bounds__(256) void concat_kernel(
    const half_t* __restrict__ x, long xObjStride, const half_t* __restrict__ h,
    half_t* __restrict__ cat, int B, int Cx, int Ch, int HW)
{
  long total = (long)B * (Cx + Ch) * HW;
  long idx   = (long)blockIdx.x * blockDim.x + threadIdx.x;
  if (idx >= total) return;
  int p = (int)(idx % HW);
  int c = (int)((idx / HW) % (Cx + Ch));
  int b = (int)(idx / ((long)(Cx + Ch) * HW));
  half_t v;
  if (c < Cx)      v = x[(long)b * xObjStride + (long)c * HW + p];
  else if (h)      v = h[((long)b * Ch + (c - Cx)) * HW + p];
  else             v = (half_t)0.0f;
  cat[idx] = v;
}

// conv-LSTM gates: cc = [i f o g] blocks over channels (f32), update c (f32),
// emit h (f16 for next conv; optionally f32 to final output).
__global__ __launch_bounds__(256) void lstm_gate_kernel(
    const float* __restrict__ cc, float* __restrict__ cst,
    half_t* __restrict__ hout, float* __restrict__ hf32,
    int B, int Ch, int HW)
{
  long total = (long)B * Ch * HW;
  long idx   = (long)blockIdx.x * blockDim.x + threadIdx.x;
  if (idx >= total) return;
  int p  = (int)(idx % HW);
  int ch = (int)((idx / HW) % Ch);
  int b  = (int)(idx / ((long)Ch * HW));
  long base = (long)b * 4 * Ch * HW + p;
  float gi = cc[base + (long)ch * HW];
  float gf = cc[base + (long)(Ch + ch) * HW];
  float go = cc[base + (long)(2 * Ch + ch) * HW];
  float gg = cc[base + (long)(3 * Ch + ch) * HW];
  float c  = sigf(gf) * cst[idx] + sigf(gi) * tanhf(gg);
  cst[idx] = c;
  float hv = sigf(go) * tanhf(c);
  hout[idx] = (half_t)hv;
  if (hf32) hf32[idx] = hv;
}

__global__ __launch_bounds__(256) void zero_f32_kernel(float* __restrict__ p, long n)
{
  long idx = (long)blockIdx.x * blockDim.x + threadIdx.x;
  if (idx < n) p[idx] = 0.f;
}

// ---------------------------------------------------------------------------
extern "C" void kernel_launch(void* const* d_in, const int* in_sizes, int n_in,
                              void* d_out, int out_size, void* d_ws, size_t ws_size,
                              hipStream_t stream)
{
  (void)in_sizes; (void)n_in; (void)out_size; (void)ws_size;

  const float* h_in = (const float*)d_in[0];
  const int*   objs = (const int*)d_in[1];
  // d_in[2] = obj_to_img (implied layout: o = b*T + t, B=4, T=2)
  const float* c1w = (const float*)d_in[3];
  const float* c2w = (const float*)d_in[4];
  const float* c3w = (const float*)d_in[5];
  const float* c4w = (const float*)d_in[6];
  const float* c5w = (const float*)d_in[7];
  const float* e1  = (const float*)d_in[8];
  const float* e2  = (const float*)d_in[9];
  const float* e3  = (const float*)d_in[10];
  const float* e4  = (const float*)d_in[11];
  const float* e5  = (const float*)d_in[12];
  const float* lw0 = (const float*)d_in[13];
  const float* lb0 = (const float*)d_in[14];
  const float* lw1 = (const float*)d_in[15];
  const float* lb1 = (const float*)d_in[16];
  const float* lw2 = (const float*)d_in[17];
  const float* lb2 = (const float*)d_in[18];

  // ---- workspace bump allocator -------------------------------------------
  char*  ws  = (char*)d_ws;
  size_t off = 0;
  auto alloc = [&](size_t bytes) -> void* {
    void* p = ws + off;
    off += (bytes + 255) & ~(size_t)255;
    return p;
  };

  half_t* wc1 = (half_t*)alloc((size_t)16384 * 2);
  half_t* wc2 = (half_t*)alloc((size_t)32768 * 2);
  half_t* wc3 = (half_t*)alloc((size_t)131072 * 2);
  half_t* wc4 = (half_t*)alloc((size_t)524288 * 2);
  half_t* wc5 = (half_t*)alloc((size_t)2097152 * 2);
  half_t* wl0 = (half_t*)alloc((size_t)8192000 * 2);
  half_t* wl1 = (half_t*)alloc((size_t)1228800 * 2);
  half_t* wl2 = (half_t*)alloc((size_t)819200 * 2);
  float*  conv_out = (float*)alloc((size_t)16777216 * 4);   // max conv output (stage 1)
  half_t* xa  = (half_t*)alloc((size_t)16777216 * 2);       // x1 / x3 / x5
  half_t* xb  = (half_t*)alloc((size_t)8388608 * 2);        // x2 / x4
  float*  meanb = (float*)alloc(512 * 4);
  float*  varb  = (float*)alloc(512 * 4);
  half_t* cat = (half_t*)alloc((size_t)655360 * 2);         // B*(512+128)*256
  float*  cc  = (float*)alloc((size_t)524288 * 4);          // B*512*256
  float*  cst = (float*)alloc((size_t)131072 * 4);          // B*128*256
  half_t* hs0 = (half_t*)alloc((size_t)262144 * 2);         // T*B*128*256
  half_t* hs1 = (half_t*)alloc((size_t)131072 * 2);         // T*B*64*256
  half_t* hs2 = (half_t*)alloc((size_t)131072 * 2);

  // ---- pack all weights into A-fragment order (fused f32->f16) ------------
  auto pack = [&](const float* wsrc, half_t* dst, int Cout, int Cin, int KH, int KW) {
    long total = (long)Cout * Cin * KH * KW / 2;
    pack_a_kernel<<<(int)((total + 255) / 256), 256, 0, stream>>>(wsrc, dst, Cout, Cin, KH, KW);
  };
  pack(c1w, wc1, 32, 512, 1, 1);
  pack(c2w, wc2, 64, 32, 4, 4);
  pack(c3w, wc3, 128, 64, 4, 4);
  pack(c4w, wc4, 256, 128, 4, 4);
  pack(c5w, wc5, 512, 256, 4, 4);
  pack(lw0, wl0, 512, 640, 5, 5);
  pack(lw1, wl1, 256, 192, 5, 5);
  pack(lw2, wl2, 256, 128, 5, 5);

  auto conv = [&](const void* in, bool f32in, const half_t* wt, const float* bias,
                  float* out, int NO, int Cin, int Hin, int Win,
                  int Cout, int Hout, int Wout, int KH, int KW, int st, int pad) {
    long blocks = (long)(Cout >> 5) * (((long)NO * Hout * Wout) >> 7);
    if (f32in)
      wmma_conv_kernel<true><<<(int)blocks, 256, 0, stream>>>(
          in, wt, bias, out, NO, Cin, Hin, Win, Cout, Hout, Wout, KH, KW, st, pad);
    else
      wmma_conv_kernel<false><<<(int)blocks, 256, 0, stream>>>(
          in, wt, bias, out, NO, Cin, Hin, Win, Cout, Hout, Wout, KH, KW, st, pad);
  };

  auto cbn = [&](const float* x, const float* e, half_t* outp,
                 int NO, int C, int HW, bool lrelu) {
    channel_stats_kernel<<<C, 256, 0, stream>>>(x, meanb, varb, NO, C, HW);
    long total = (long)NO * C * HW;
    cbn_apply_kernel<<<(int)((total + 255) / 256), 256, 0, stream>>>(
        x, meanb, varb, e, objs, outp, C, HW, total, lrelu ? 1 : 0);
  };

  // ---- encoder pyramid ----------------------------------------------------
  conv(h_in, true, wc1, nullptr, conv_out, 8, 512, 254, 254, 32, 256, 256, 1, 1, 1, 1);
  cbn(conv_out, e1, xa, 8, 32, 65536, true);
  conv(xa, false, wc2, nullptr, conv_out, 8, 32, 256, 256, 64, 128, 128, 4, 4, 2, 1);
  cbn(conv_out, e2, xb, 8, 64, 16384, true);
  conv(xb, false, wc3, nullptr, conv_out, 8, 64, 128, 128, 128, 64, 64, 4, 4, 2, 1);
  cbn(conv_out, e3, xa, 8, 128, 4096, true);
  conv(xa, false, wc4, nullptr, conv_out, 8, 128, 64, 64, 256, 32, 32, 4, 4, 2, 1);
  cbn(conv_out, e4, xb, 8, 256, 1024, false);
  conv(xb, false, wc5, nullptr, conv_out, 8, 256, 32, 32, 512, 16, 16, 4, 4, 2, 1);
  cbn(conv_out, e5, xa, 8, 512, 256, false);

  // ---- conv-LSTM stack ----------------------------------------------------
  const int B = 4, T = 2, HW = 256;
  auto runLayer = [&](const half_t* xbase, long xTstride, long xObjStride,
                      int Cx, int Ch, const half_t* w, const float* bias,
                      half_t* hs, float* finalOut) {
    long cstN = (long)B * Ch * HW;
    zero_f32_kernel<<<(int)((cstN + 255) / 256), 256, 0, stream>>>(cst, cstN);
    for (int t = 0; t < T; ++t) {
      const half_t* hprev = (t == 0) ? nullptr : hs + (long)(t - 1) * B * Ch * HW;
      long catN = (long)B * (Cx + Ch) * HW;
      concat_kernel<<<(int)((catN + 255) / 256), 256, 0, stream>>>(
          xbase + (long)t * xTstride, xObjStride, hprev, cat, B, Cx, Ch, HW);
      conv(cat, false, w, bias, cc, B, Cx + Ch, 16, 16, 4 * Ch, 16, 16, 5, 5, 1, 2);
      float* hf32 = (finalOut && t == T - 1) ? finalOut : nullptr;
      lstm_gate_kernel<<<(int)((cstN + 255) / 256), 256, 0, stream>>>(
          cc, cst, hs + (long)t * B * Ch * HW, hf32, B, Ch, HW);
    }
  };

  // layer 0: x = x5 (object o = b*T + t), stride between b's = T*512*HW
  runLayer(xa,  (long)512 * HW, (long)T * 512 * HW, 512, 128, wl0, lb0, hs0, nullptr);
  // layer 1: x = hs0 (t, b, 128, HW)
  runLayer(hs0, (long)B * 128 * HW, (long)128 * HW, 128, 64, wl1, lb1, hs1, nullptr);
  // layer 2: x = hs1; final h written f32 to d_out
  runLayer(hs1, (long)B * 64 * HW, (long)64 * HW, 64, 64, wl2, lb2, hs2, (float*)d_out);
}
